// MidLevelLSTM_peephole_14207751815713
// MI455X (gfx1250) — compile-verified, matrix operand order checked by reading, not run on hardware
//
#include <hip/hip_runtime.h>
#include <hip/hip_bf16.h>

// Problem dims
#define XD 1024
#define HD 2048
#define YD 1024
#define TT 1024

typedef __attribute__((ext_vector_type(16))) __bf16 v16bf;
typedef __attribute__((ext_vector_type(8)))  float  v8f;

union FragBF { uint4 q[2]; v16bf v; };
static_assert(sizeof(FragBF) == 32, "frag size");

__device__ __forceinline__ unsigned short f2bf(float f) {
  unsigned u = __float_as_uint(f);
  unsigned r = u + 0x7fffu + ((u >> 16) & 1u);   // round-to-nearest-even
  return (unsigned short)(r >> 16);
}

__device__ __forceinline__ void bf2x(unsigned u, float& lo, float& hi) {
  lo = __uint_as_float(u << 16);
  hi = __uint_as_float(u & 0xffff0000u);
}

// ---------------------------------------------------------------------------
// fp32 [rows x src_ld] column-slice (col0..col0+cols) -> contiguous bf16 [rows x cols]
// ---------------------------------------------------------------------------
__global__ __launch_bounds__(256) void conv_slice(const float* __restrict__ src,
                                                  int src_ld, int col0,
                                                  unsigned short* __restrict__ dst,
                                                  int rows, int cols) {
  size_t n = (size_t)rows * cols;
  for (size_t i = (size_t)blockIdx.x * blockDim.x + threadIdx.x; i < n;
       i += (size_t)gridDim.x * blockDim.x) {
    size_t r = i / (size_t)cols, c = i - r * (size_t)cols;
    dst[i] = f2bf(src[r * (size_t)src_ld + col0 + c]);
  }
}

// ---------------------------------------------------------------------------
// bf16 NT GEMM via WMMA:  C(m,n) = sum_k A[m][k] * Bt[n][k] + bias[m]
//   A : [M][K] row-major (lda), Bt : [N][K] row-major (ldb)
//   C stored at C[n*ldc + m]  (time-major output for both Gx and ys)
// Block = 8 waves; wave tile 32x32 (2x2 WMMA 16x16x32), block tile 64x128.
// ---------------------------------------------------------------------------
__global__ __launch_bounds__(256) void gemm_bf16_nt(const unsigned short* __restrict__ A,
                                                    const unsigned short* __restrict__ Bt,
                                                    float* __restrict__ C,
                                                    const float* __restrict__ bias,
                                                    int M, int N, int K,
                                                    int lda, int ldb, int ldc) {
  const int lane = threadIdx.x & 31;
  const int wave = threadIdx.x >> 5;
  const int wm = wave & 1;        // 2 waves along M
  const int wn = wave >> 1;       // 4 waves along N
  const int m0 = blockIdx.x * 64 + wm * 32;
  const int n0 = blockIdx.y * 128 + wn * 32;
  const int l15   = lane & 15;
  const int khalf = (lane >> 4) * 8;   // ISA 16-bit A/B layout: lanes 16..31 hold K+8 / K+24

  v8f acc00 = {}, acc01 = {}, acc10 = {}, acc11 = {};

  const unsigned short* a0p = A  + (size_t)(m0 + l15)      * lda + khalf;
  const unsigned short* a1p = A  + (size_t)(m0 + 16 + l15) * lda + khalf;
  const unsigned short* b0p = Bt + (size_t)(n0 + l15)      * ldb + khalf;
  const unsigned short* b1p = Bt + (size_t)(n0 + 16 + l15) * ldb + khalf;

  for (int k0 = 0; k0 < K; k0 += 32) {
    FragBF a0, a1, b0, b1;
    a0.q[0] = *(const uint4*)(a0p + k0);      a0.q[1] = *(const uint4*)(a0p + k0 + 16);
    a1.q[0] = *(const uint4*)(a1p + k0);      a1.q[1] = *(const uint4*)(a1p + k0 + 16);
    b0.q[0] = *(const uint4*)(b0p + k0);      b0.q[1] = *(const uint4*)(b0p + k0 + 16);
    b1.q[0] = *(const uint4*)(b1p + k0);      b1.q[1] = *(const uint4*)(b1p + k0 + 16);

    acc00 = __builtin_amdgcn_wmma_f32_16x16x32_bf16(false, a0.v, false, b0.v, (short)0, acc00, false, false);
    acc01 = __builtin_amdgcn_wmma_f32_16x16x32_bf16(false, a0.v, false, b1.v, (short)0, acc01, false, false);
    acc10 = __builtin_amdgcn_wmma_f32_16x16x32_bf16(false, a1.v, false, b0.v, (short)0, acc10, false, false);
    acc11 = __builtin_amdgcn_wmma_f32_16x16x32_bf16(false, a1.v, false, b1.v, (short)0, acc11, false, false);
  }

  // C/D layout: VGPR j -> M = tile_m + (lane>=16 ? 8 : 0) + j, N = tile_n + (lane&15)
  const int mo = (lane >> 4) * 8;
  {
    v8f* accs[4] = {&acc00, &acc01, &acc10, &acc11};
    #pragma unroll
    for (int tmn = 0; tmn < 4; ++tmn) {
      int mt = tmn >> 1, nt = tmn & 1;
      int m = m0 + mt * 16 + mo;
      int n = n0 + nt * 16 + l15;
      float* p = C + (size_t)n * ldc + m;
      v8f a = *accs[tmn];
      #pragma unroll
      for (int j = 0; j < 8; ++j) p[j] = a[j] + bias[m + j];
    }
  }
}

// ---------------------------------------------------------------------------
// One LSTM timestep (sequential recurrence).  L2-BW bound: streams 7 bf16
// 2048x2048 matrices (58.7 MB, L2-resident).  Block = 256 threads = 8 waves,
// grid = 256 blocks; block handles 8 h-rows, each wave does 7 of the 56
// (row x matrix) dot products, wave32 shuffle reduction.
// a_prev / c_prev are staged into LDS with GLOBAL_LOAD_ASYNC_TO_LDS_B128
// (ASYNCcnt path, no VGPR roundtrip), completed with s_wait_asynccnt 0.
// Wrec matrix order: 0=Wc_a 1=Wu_a 2=Wf_a 3=Wo_a 4=Wu_c 5=Wf_c 6=Wo_c
// Gx*_t already contains (W*_x @ x_t + b*).
// ---------------------------------------------------------------------------
__global__ __launch_bounds__(256) void lstm_step(const float* __restrict__ a_prev,
                                                 const float* __restrict__ c_prev,
                                                 const unsigned short* __restrict__ Wrec,
                                                 const float* __restrict__ Gxc_t,
                                                 const float* __restrict__ Gxu_t,
                                                 const float* __restrict__ Gxf_t,
                                                 const float* __restrict__ Gxo_t,
                                                 float* __restrict__ a_out,
                                                 float* __restrict__ c_out,
                                                 unsigned short* __restrict__ a_bf_out) {
  __shared__ float sh_a[HD];
  __shared__ float sh_c[HD];
  __shared__ float part[8][7];

  // ---- async DMA of a_prev/c_prev (8 KB each) into LDS -------------------
  // GV mode: per-lane 64-bit global address, per-lane LDS byte offset (the
  // low 32 bits of a generic pointer to LDS are the LDS offset, ISA 10.2).
  #pragma unroll
  for (int it = 0; it < 2; ++it) {
    int i = (it * 256 + threadIdx.x) * 4;            // float index, 16B chunks
    unsigned la = (unsigned)(uintptr_t)(&sh_a[i]);
    unsigned lc = (unsigned)(uintptr_t)(&sh_c[i]);
    const float* ga = a_prev + i;
    const float* gc = c_prev + i;
    asm volatile("global_load_async_to_lds_b128 %0, %1, off"
                 :: "v"(la), "v"(ga) : "memory");
    asm volatile("global_load_async_to_lds_b128 %0, %1, off"
                 :: "v"(lc), "v"(gc) : "memory");
  }
  asm volatile("s_wait_asynccnt 0x0" ::: "memory");
  __syncthreads();

  const int lane = threadIdx.x & 31;
  const int wave = threadIdx.x >> 5;
  const int rbase = blockIdx.x * 8;

  for (int task = wave; task < 56; task += 8) {
    const int mat = task % 7;
    const int rl  = task / 7;
    const int row = rbase + rl;
    const unsigned short* w = Wrec + (size_t)mat * HD * HD + (size_t)row * HD;
    const float* vec = (mat < 4) ? sh_a : sh_c;

    // prefetch the weight row of this wave's next task into cache
    int ntask = task + 8;
    if (ntask < 56) {
      const unsigned short* wn = Wrec + (size_t)(ntask % 7) * HD * HD +
                                 (size_t)(rbase + ntask / 7) * HD;
      __builtin_prefetch(wn + lane * 8, 0, 0);
    }

    float sum = 0.f;
    #pragma unroll
    for (int kk = 0; kk < 8; ++kk) {
      int k = kk * 256 + lane * 8;
      uint4 q = *(const uint4*)(w + k);
      float l0, h0, l1, h1, l2, h2, l3, h3;
      bf2x(q.x, l0, h0); bf2x(q.y, l1, h1); bf2x(q.z, l2, h2); bf2x(q.w, l3, h3);
      sum += l0 * vec[k + 0] + h0 * vec[k + 1] + l1 * vec[k + 2] + h1 * vec[k + 3]
           + l2 * vec[k + 4] + h2 * vec[k + 5] + l3 * vec[k + 6] + h3 * vec[k + 7];
    }
    #pragma unroll
    for (int off = 16; off > 0; off >>= 1) sum += __shfl_xor(sum, off, 32);
    if (lane == 0) part[rl][mat] = sum;
  }
  __syncthreads();

  if (threadIdx.x < 8) {
    const int rl = threadIdx.x;
    const int row = rbase + rl;
    float pre_c = part[rl][0] + Gxc_t[row];
    float pre_u = part[rl][1] + part[rl][4] + Gxu_t[row];
    float pre_f = part[rl][2] + part[rl][5] + Gxf_t[row];
    float pre_o = part[rl][3] + part[rl][6] + Gxo_t[row];
    float ct = tanhf(pre_c);
    float gu = 1.f / (1.f + __expf(-pre_u));
    float gf = 1.f / (1.f + __expf(-pre_f));
    float go = 1.f / (1.f + __expf(-pre_o));
    float c_new = ct * gu + sh_c[row] * gf;
    float a_new = go * tanhf(c_new);
    c_out[row] = c_new;
    a_out[row] = a_new;
    a_bf_out[row] = f2bf(a_new);
  }
}

__global__ __launch_bounds__(256) void copy_ac(const float* __restrict__ a,
                                               const float* __restrict__ c,
                                               float* __restrict__ out) {
  int i = blockIdx.x * blockDim.x + threadIdx.x;
  if (i < HD) out[i] = a[i];
  else if (i < 2 * HD) out[i] = c[i - HD];
}

// ---------------------------------------------------------------------------
extern "C" void kernel_launch(void* const* d_in, const int* in_sizes, int n_in,
                              void* d_out, int out_size, void* d_ws, size_t ws_size,
                              hipStream_t stream) {
  (void)in_sizes; (void)n_in; (void)out_size; (void)ws_size;
  const float* x   = (const float*)d_in[0];
  const float* h_a = (const float*)d_in[1];
  const float* h_c = (const float*)d_in[2];
  const float* Wc  = (const float*)d_in[3];
  const float* bc  = (const float*)d_in[4];
  const float* Wu  = (const float*)d_in[5];
  const float* bu  = (const float*)d_in[6];
  const float* Wf  = (const float*)d_in[7];
  const float* bfv = (const float*)d_in[8];
  const float* Wo  = (const float*)d_in[9];
  const float* bo  = (const float*)d_in[10];
  const float* Wy  = (const float*)d_in[11];
  const float* by  = (const float*)d_in[12];

  // workspace carve (256B aligned partitions)
  char* p = (char*)d_ws;
  auto carve = [&](size_t bytes) -> void* {
    void* r = (void*)p;
    p += (bytes + 255) & ~(size_t)255;
    return r;
  };
  unsigned short* Wrec = (unsigned short*)carve(7ull * HD * HD * 2);      // 58.7 MB
  unsigned short* Wx   = (unsigned short*)carve(4ull * HD * XD * 2);      // 16.8 MB
  unsigned short* Wyb  = (unsigned short*)carve((size_t)YD * HD * 2);     //  4.2 MB
  unsigned short* Xb   = (unsigned short*)carve((size_t)TT * XD * 2);     //  2.1 MB
  float*          Gx   = (float*)carve(4ull * TT * HD * 4);               // 33.6 MB
  unsigned short* Ah   = (unsigned short*)carve((size_t)TT * HD * 2);     //  4.2 MB
  float*          abuf = (float*)carve(2ull * HD * 4);
  float*          cbuf = (float*)carve(2ull * HD * 4);

  const dim3 cb(256);
  auto cg = [](size_t n) { return dim3((unsigned)((n + 255) / 256)); };
  const size_t HH = (size_t)HD * HD, HX = (size_t)HD * XD;

  // --- Phase A: fp32 -> bf16 weight slices (ax = [a, x]; axc = [a, x, c]) ---
  conv_slice<<<cg(HH), cb, 0, stream>>>(Wc, XD + HD,     0,    Wrec + 0 * HH, HD, HD);   // Wc_a
  conv_slice<<<cg(HH), cb, 0, stream>>>(Wu, XD + 2 * HD, 0,    Wrec + 1 * HH, HD, HD);   // Wu_a
  conv_slice<<<cg(HH), cb, 0, stream>>>(Wf, XD + 2 * HD, 0,    Wrec + 2 * HH, HD, HD);   // Wf_a
  conv_slice<<<cg(HH), cb, 0, stream>>>(Wo, XD + 2 * HD, 0,    Wrec + 3 * HH, HD, HD);   // Wo_a
  conv_slice<<<cg(HH), cb, 0, stream>>>(Wu, XD + 2 * HD, HD + XD, Wrec + 4 * HH, HD, HD); // Wu_c
  conv_slice<<<cg(HH), cb, 0, stream>>>(Wf, XD + 2 * HD, HD + XD, Wrec + 5 * HH, HD, HD); // Wf_c
  conv_slice<<<cg(HH), cb, 0, stream>>>(Wo, XD + 2 * HD, HD + XD, Wrec + 6 * HH, HD, HD); // Wo_c
  conv_slice<<<cg(HX), cb, 0, stream>>>(Wc, XD + HD,     HD, Wx + 0 * HX, HD, XD);       // Wc_x
  conv_slice<<<cg(HX), cb, 0, stream>>>(Wu, XD + 2 * HD, HD, Wx + 1 * HX, HD, XD);       // Wu_x
  conv_slice<<<cg(HX), cb, 0, stream>>>(Wf, XD + 2 * HD, HD, Wx + 2 * HX, HD, XD);       // Wf_x
  conv_slice<<<cg(HX), cb, 0, stream>>>(Wo, XD + 2 * HD, HD, Wx + 3 * HX, HD, XD);       // Wo_x
  conv_slice<<<cg((size_t)YD * HD), cb, 0, stream>>>(Wy, HD, 0, Wyb, YD, HD);
  conv_slice<<<cg((size_t)TT * XD), cb, 0, stream>>>(x,  XD, 0, Xb,  TT, XD);

  // --- Phase B: batched x-projections via WMMA  Gx_g[t][h] = W_gx @ x_t + b_g ---
  const float* gbias[4] = {bc, bu, bfv, bo};
  for (int g = 0; g < 4; ++g) {
    gemm_bf16_nt<<<dim3(HD / 64, TT / 128), 256, 0, stream>>>(
        Wx + (size_t)g * HX, Xb, Gx + (size_t)g * TT * HD, gbias[g],
        HD, TT, XD, XD, XD, HD);
  }

  // --- Phase C: sequential recurrence (one launch per timestep) ---
  for (int t = 0; t < TT; ++t) {
    const float* ap = t ? abuf + ((size_t)((t - 1) & 1)) * HD : h_a;
    const float* cp = t ? cbuf + ((size_t)((t - 1) & 1)) * HD : h_c;
    float* an = abuf + (size_t)(t & 1) * HD;
    float* cn = cbuf + (size_t)(t & 1) * HD;
    lstm_step<<<dim3(HD / 8), 256, 0, stream>>>(
        ap, cp, Wrec,
        Gx + 0ull * TT * HD + (size_t)t * HD,
        Gx + 1ull * TT * HD + (size_t)t * HD,
        Gx + 2ull * TT * HD + (size_t)t * HD,
        Gx + 3ull * TT * HD + (size_t)t * HD,
        an, cn, Ah + (size_t)t * HD);
  }

  // --- Phase D: batched output projection via WMMA  ys[t][y] = Wy @ a_t + by ---
  gemm_bf16_nt<<<dim3(YD / 64, TT / 128), 256, 0, stream>>>(
      Wyb, Ah, (float*)d_out, by, YD, TT, HD, HD, HD, YD);

  // --- final (a, c) appended after ys ---
  copy_ac<<<dim3(16), 256, 0, stream>>>(abuf + (size_t)((TT - 1) & 1) * HD,
                                        cbuf + (size_t)((TT - 1) & 1) * HD,
                                        (float*)d_out + (size_t)TT * YD);
}
